// GCNN_with_descriptors_25872882991627
// MI455X (gfx1250) — compile-verified, matrix operand order checked by reading, not run on hardware
//
#include <hip/hip_runtime.h>

// ---------------- problem constants ----------------
#define Nn   20000
#define Ec   80000
#define Bc   32
#define FPc  1024
#define ODc  128
#define Lc   384
#define DDc  80
#define TDc  63
#define Dc   64      // TD+1
#define Hh   4
#define HDc  16
#define FFc  128
#define Sc   768     // 2*L
#define NEGV (-1000000000.0f)
#define LDA  40      // LDS row stride in bf16 elems (80B, 8B-aligned, conflict-skewed)

typedef __attribute__((ext_vector_type(16))) __bf16 v16bf;
typedef __attribute__((ext_vector_type(4)))  __bf16 v4bf;
typedef __attribute__((ext_vector_type(8)))  float  v8f;

__device__ __forceinline__ float leakyf(float x) { return x >= 0.f ? x : 0.01f * x; }

// ======================================================================
// 1) WMMA bf16 GEMM:  H[nrows x 1024] = X[nrows x 1024] @ W^T
//    block tile 128(M) x 128(N), 8 waves as 4(M) x 2(N),
//    each wave: 2 M-subtiles x 4 N-subtiles = 8 accumulators,
//    K step 32, double-buffered LDS, branch-free clamped loads.
// ======================================================================
__global__ __launch_bounds__(256) void k_gemm_bf16(const float* __restrict__ X,
                                                   const float* __restrict__ W,
                                                   float* __restrict__ Hout, int nrows) {
  __shared__ __bf16 As[2][128 * LDA];
  __shared__ __bf16 Bs[2][128 * LDA];
  const int tid  = threadIdx.x;
  const int wave = tid >> 5;
  const int lane = tid & 31;
  const int g    = lane >> 4;
  const int r    = lane & 15;
  const int wm   = wave >> 1;     // 0..3
  const int wn   = wave & 1;      // 0..1
  const int mBase = blockIdx.x * 128;
  const int nBase = blockIdx.y * 128;

  v8f acc[2][4];
#pragma unroll
  for (int mi = 0; mi < 2; mi++)
#pragma unroll
    for (int ni = 0; ni < 4; ni++) acc[mi][ni] = (v8f){};

  float4 ra[4], rb[4];

  auto gload = [&](int kt) {
#pragma unroll
    for (int i = 0; i < 4; i++) {
      int e  = tid + i * 256;          // float4 element index over 128x8
      int rw = e >> 3;
      int kq = (e & 7) * 4;
      int grow = mBase + rw;
      if (grow > nrows - 1) grow = nrows - 1;   // clamp: branch-free OOB handling
      ra[i] = *(const float4*)(X + (size_t)grow * FPc + kt + kq);
      rb[i] = *(const float4*)(W + (size_t)(nBase + rw) * FPc + kt + kq);
    }
  };
  auto lstore = [&](int buf) {
#pragma unroll
    for (int i = 0; i < 4; i++) {
      int e  = tid + i * 256;
      int rw = e >> 3;
      int kq = (e & 7) * 4;
      v4bf ta = { (__bf16)ra[i].x, (__bf16)ra[i].y, (__bf16)ra[i].z, (__bf16)ra[i].w };
      v4bf tb = { (__bf16)rb[i].x, (__bf16)rb[i].y, (__bf16)rb[i].z, (__bf16)rb[i].w };
      *(v4bf*)&As[buf][rw * LDA + kq] = ta;
      *(v4bf*)&Bs[buf][rw * LDA + kq] = tb;
    }
  };

  gload(0);
  lstore(0);
  __syncthreads();

  int buf = 0;
  for (int kt = 0; kt < FPc; kt += 32) {
    const bool more = (kt + 32) < FPc;
    if (more) gload(kt + 32);   // issue next tile's global loads early

    v16bf a0, a1, b0, b1, b2, b3;
#pragma unroll
    for (int e2 = 0; e2 < 16; e2++) {
      int k = e2 + 8 * g + ((e2 >= 8) ? 8 : 0);
      a0[e2] = As[buf][(wm * 32 +  0 + r) * LDA + k];
      a1[e2] = As[buf][(wm * 32 + 16 + r) * LDA + k];
      b0[e2] = Bs[buf][(wn * 64 +  0 + r) * LDA + k];
      b1[e2] = Bs[buf][(wn * 64 + 16 + r) * LDA + k];
      b2[e2] = Bs[buf][(wn * 64 + 32 + r) * LDA + k];
      b3[e2] = Bs[buf][(wn * 64 + 48 + r) * LDA + k];
    }
    acc[0][0] = __builtin_amdgcn_wmma_f32_16x16x32_bf16(false, a0, false, b0, (short)0, acc[0][0], false, false);
    acc[0][1] = __builtin_amdgcn_wmma_f32_16x16x32_bf16(false, a0, false, b1, (short)0, acc[0][1], false, false);
    acc[0][2] = __builtin_amdgcn_wmma_f32_16x16x32_bf16(false, a0, false, b2, (short)0, acc[0][2], false, false);
    acc[0][3] = __builtin_amdgcn_wmma_f32_16x16x32_bf16(false, a0, false, b3, (short)0, acc[0][3], false, false);
    acc[1][0] = __builtin_amdgcn_wmma_f32_16x16x32_bf16(false, a1, false, b0, (short)0, acc[1][0], false, false);
    acc[1][1] = __builtin_amdgcn_wmma_f32_16x16x32_bf16(false, a1, false, b1, (short)0, acc[1][1], false, false);
    acc[1][2] = __builtin_amdgcn_wmma_f32_16x16x32_bf16(false, a1, false, b2, (short)0, acc[1][2], false, false);
    acc[1][3] = __builtin_amdgcn_wmma_f32_16x16x32_bf16(false, a1, false, b3, (short)0, acc[1][3], false, false);

    if (more) lstore(buf ^ 1);  // write next tile into the other buffer
    __syncthreads();
    buf ^= 1;
  }

#pragma unroll
  for (int mi = 0; mi < 2; mi++)
#pragma unroll
    for (int ni = 0; ni < 4; ni++)
#pragma unroll
      for (int rr = 0; rr < 8; rr++) {
        int row = mBase + wm * 32 + mi * 16 + rr + 8 * g;
        if (row < nrows)
          Hout[(size_t)row * FPc + nBase + wn * 64 + ni * 16 + r] = acc[mi][ni][rr];
      }
}

// ======================================================================
// 2) GCN aggregation
// ======================================================================
__global__ void k_deg_init(float* deg, int n) {
  int i = blockIdx.x * blockDim.x + threadIdx.x;
  if (i < n) deg[i] = 1.0f;
}
__global__ void k_deg_edges(const int* __restrict__ ei, float* deg, int nE) {
  int e = blockIdx.x * blockDim.x + threadIdx.x;
  if (e < nE) atomicAdd(&deg[ei[nE + e]], 1.0f);
}
__global__ void k_dinv(float* deg, int n) {
  int i = blockIdx.x * blockDim.x + threadIdx.x;
  if (i < n) { float d = deg[i]; deg[i] = d > 0.f ? rsqrtf(d) : 0.f; }
}
__global__ void k_selfbias(const float* __restrict__ Hm, const float* __restrict__ dinv,
                           const float* __restrict__ bias, float* __restrict__ agg, int n) {
  long long idx = (long long)blockIdx.x * blockDim.x + threadIdx.x;
  if (idx >= (long long)n * FPc) return;
  int i = (int)(idx >> 10), f = (int)(idx & 1023);
  float di = dinv[i];
  agg[idx] = bias[f] + Hm[idx] * di * di;
}
__global__ __launch_bounds__(256) void k_edge_scatter(const int* __restrict__ ei,
                                                      const float* __restrict__ Hm,
                                                      const float* __restrict__ dinv,
                                                      float* __restrict__ agg, int nE) {
  int e = blockIdx.x;
  if (e >= nE) return;
  int s = ei[e], d = ei[nE + e];
  float c = dinv[s] * dinv[d];
  const float4* hs = (const float4*)(Hm + (size_t)s * FPc);
  float4 v = hs[threadIdx.x];
  float* out = agg + (size_t)d * FPc + threadIdx.x * 4;
  atomicAdd(out + 0, v.x * c);
  atomicAdd(out + 1, v.y * c);
  atomicAdd(out + 2, v.z * c);
  atomicAdd(out + 3, v.w * c);
}
__global__ void k_pool_zero(float* ps, float* pc) {
  int i = blockIdx.x * blockDim.x + threadIdx.x;
  if (i < Bc * FPc) ps[i] = 0.f;
  if (i < Bc) pc[i] = 0.f;
}
// leaky + segment-mean numerator: 16 consecutive (batch-sorted) rows per block,
// register accumulation, atomic flush only on batch boundary.
__global__ __launch_bounds__(256) void k_pool(const float* __restrict__ agg,
                                              const int* __restrict__ batch,
                                              float* __restrict__ ps, float* __restrict__ pc, int n) {
  int base = blockIdx.x * 16;
  if (base >= n) return;
  int tid = threadIdx.x;
  int cur = batch[base];
  float4 acc = {0.f, 0.f, 0.f, 0.f};
  float cnt = 0.f;
  for (int j = 0; j < 16; j++) {
    int i = base + j;
    if (i >= n) break;
    int b = batch[i];
    if (b != cur) {
      float* o = ps + (size_t)cur * FPc + tid * 4;
      atomicAdd(o + 0, acc.x); atomicAdd(o + 1, acc.y);
      atomicAdd(o + 2, acc.z); atomicAdd(o + 3, acc.w);
      if (tid == 0) atomicAdd(&pc[cur], cnt);
      acc = (float4){0.f, 0.f, 0.f, 0.f}; cnt = 0.f; cur = b;
    }
    float4 v = ((const float4*)(agg + (size_t)i * FPc))[tid];
    acc.x += leakyf(v.x); acc.y += leakyf(v.y);
    acc.z += leakyf(v.z); acc.w += leakyf(v.w);
    cnt += 1.f;
  }
  float* o = ps + (size_t)cur * FPc + tid * 4;
  atomicAdd(o + 0, acc.x); atomicAdd(o + 1, acc.y);
  atomicAdd(o + 2, acc.z); atomicAdd(o + 3, acc.w);
  if (tid == 0) atomicAdd(&pc[cur], cnt);
}
__global__ __launch_bounds__(256) void k_fc(const float* __restrict__ ps, const float* __restrict__ pc,
                                            const float* __restrict__ fcw, const float* __restrict__ fcb,
                                            float* __restrict__ xg) {
  __shared__ float red[256];
  int b = blockIdx.x / ODc, o = blockIdx.x % ODc;
  float inv = 1.0f / fmaxf(pc[b], 1.0f);
  float acc = 0.f;
  for (int k = threadIdx.x; k < FPc; k += 256)
    acc += ps[(size_t)b * FPc + k] * inv * fcw[(size_t)o * FPc + k];
  red[threadIdx.x] = acc;
  __syncthreads();
  for (int st = 128; st > 0; st >>= 1) {
    if (threadIdx.x < st) red[threadIdx.x] += red[threadIdx.x + st];
    __syncthreads();
  }
  if (threadIdx.x == 0) xg[b * ODc + o] = leakyf(red[0] + fcb[o]);
}

// ======================================================================
// 3) sequence build
// ======================================================================
__global__ __launch_bounds__(64) void k_buildseq(const float* __restrict__ straight,
                                                 const float* __restrict__ flipped,
                                                 const int* __restrict__ slen, const int* __restrict__ flen,
                                                 const float* __restrict__ rw, const float* __restrict__ rb,
                                                 float* __restrict__ seq, int* __restrict__ maskb) {
  __shared__ float srow[DDc];
  int b = blockIdx.x / Sc, pos = blockIdx.x % Sc;
  int sl = slen[b], fl = flen[b];
  bool in_s = pos < sl;
  bool in_f = (pos >= sl) && (pos < sl + fl);
  const float* src = nullptr;
  if (in_s)      src = straight + ((size_t)b * Lc + min(pos, Lc - 1)) * DDc;
  else if (in_f) src = flipped  + ((size_t)b * Lc + min(max(pos - sl, 0), Lc - 1)) * DDc;
  for (int t = threadIdx.x; t < DDc; t += 64) srow[t] = src ? src[t] : 0.f;
  __syncthreads();
  int d = threadIdx.x;
  float val;
  if (d == TDc) {
    val = in_s ? 1.f : 0.f;
  } else if (src) {
    float acc = rb[d];
    const float* wr = rw + (size_t)d * DDc;
    for (int k = 0; k < DDc; k++) acc += srow[k] * wr[k];
    val = acc;
  } else {
    val = 0.f;
  }
  seq[((size_t)b * Sc + pos) * Dc + d] = val;
  if (d == 0) maskb[b * Sc + pos] = (in_s || in_f) ? 1 : 0;
}

// ======================================================================
// 4) transformer
// ======================================================================
__global__ __launch_bounds__(192) void k_qkv(const float* __restrict__ x,
                                             const float* __restrict__ qw, const float* __restrict__ qb,
                                             float* __restrict__ qkv) {
  __shared__ float xrow[Dc];
  size_t row = blockIdx.x;
  if (threadIdx.x < Dc) xrow[threadIdx.x] = x[row * Dc + threadIdx.x];
  __syncthreads();
  int j = threadIdx.x;
  float acc = qb[j];
  const float* wr = qw + (size_t)j * Dc;
  for (int k = 0; k < Dc; k++) acc += xrow[k] * wr[k];
  qkv[row * (3 * Dc) + j] = acc;
}

__global__ __launch_bounds__(256) void k_attn(const float* __restrict__ qkv,
                                              const int* __restrict__ maskb,
                                              float* __restrict__ attno) {
  __shared__ float sc[Sc];
  __shared__ float qv[HDc];
  __shared__ float outacc[HDc];
  __shared__ float red[256];
  int q = blockIdx.x % Sc;
  int h = (blockIdx.x / Sc) % Hh;
  int b = blockIdx.x / (Sc * Hh);
  int tid = threadIdx.x;
  const float* qrow = qkv + ((size_t)b * Sc + q) * (3 * Dc) + h * HDc;
  if (tid < HDc) { qv[tid] = qrow[tid]; outacc[tid] = 0.f; }
  __syncthreads();
  float mloc = -3.4e38f;
  for (int k = tid; k < Sc; k += 256) {
    float s;
    if (maskb[b * Sc + k]) {
      const float* kr = qkv + ((size_t)b * Sc + k) * (3 * Dc) + Dc + h * HDc;
      float acc = 0.f;
#pragma unroll
      for (int d = 0; d < HDc; d++) acc += qv[d] * kr[d];
      s = acc * 0.25f;
    } else s = NEGV;
    sc[k] = s;
    mloc = fmaxf(mloc, s);
  }
  red[tid] = mloc;
  __syncthreads();
  for (int st = 128; st > 0; st >>= 1) {
    if (tid < st) red[tid] = fmaxf(red[tid], red[tid + st]);
    __syncthreads();
  }
  float mx = red[0];
  __syncthreads();
  float sloc = 0.f;
  for (int k = tid; k < Sc; k += 256) {
    float e = __expf(sc[k] - mx);
    sc[k] = e;
    sloc += e;
  }
  red[tid] = sloc;
  __syncthreads();
  for (int st = 128; st > 0; st >>= 1) {
    if (tid < st) red[tid] += red[tid + st];
    __syncthreads();
  }
  float inv = 1.0f / red[0];
  int d = tid & 15, ks = tid >> 4;
  float acc = 0.f;
  for (int k = ks; k < Sc; k += 16) {
    const float* vr = qkv + ((size_t)b * Sc + k) * (3 * Dc) + 2 * Dc + h * HDc;
    acc += sc[k] * vr[d];
  }
  atomicAdd(&outacc[d], acc);
  __syncthreads();
  if (tid < HDc) attno[((size_t)b * Sc + q) * Dc + h * HDc + tid] = outacc[tid] * inv;
}

__global__ __launch_bounds__(64) void k_proj_ln(const float* __restrict__ o, const float* __restrict__ x,
                                                const float* __restrict__ ow, const float* __restrict__ ob,
                                                const float* __restrict__ lw, const float* __restrict__ lb,
                                                float* __restrict__ y) {
  __shared__ float orow[Dc];
  __shared__ float red[Dc];
  size_t row = blockIdx.x;
  int d = threadIdx.x;
  orow[d] = o[row * Dc + d];
  __syncthreads();
  float acc = ob[d];
  const float* wr = ow + (size_t)d * Dc;
  for (int k = 0; k < Dc; k++) acc += orow[k] * wr[k];
  float val = acc + x[row * Dc + d];
  red[d] = val;
  __syncthreads();
  for (int st = 32; st > 0; st >>= 1) {
    if (d < st) red[d] += red[d + st];
    __syncthreads();
  }
  float m = red[0] / Dc;
  __syncthreads();
  float diff = val - m;
  red[d] = diff * diff;
  __syncthreads();
  for (int st = 32; st > 0; st >>= 1) {
    if (d < st) red[d] += red[d + st];
    __syncthreads();
  }
  float v = red[0] / Dc;
  y[row * Dc + d] = diff * rsqrtf(v + 1e-5f) * lw[d] + lb[d];
}

__global__ __launch_bounds__(128) void k_ffn_ln(const float* __restrict__ y,
                                                const float* __restrict__ f1w, const float* __restrict__ f1b,
                                                const float* __restrict__ f2w, const float* __restrict__ f2b,
                                                const float* __restrict__ lw, const float* __restrict__ lb,
                                                float* __restrict__ x) {
  __shared__ float yrow[Dc];
  __shared__ float hbuf[FFc];
  __shared__ float red[128];
  size_t row = blockIdx.x;
  int t = threadIdx.x;
  if (t < Dc) yrow[t] = y[row * Dc + t];
  __syncthreads();
  {
    float acc = f1b[t];
    const float* wr = f1w + (size_t)t * Dc;
    for (int k = 0; k < Dc; k++) acc += yrow[k] * wr[k];
    hbuf[t] = fmaxf(acc, 0.f);
  }
  __syncthreads();
  float val = 0.f;
  if (t < Dc) {
    float acc = f2b[t];
    const float* wr = f2w + (size_t)t * FFc;
    for (int k = 0; k < FFc; k++) acc += hbuf[k] * wr[k];
    val = acc + yrow[t];
  }
  red[t] = (t < Dc) ? val : 0.f;
  __syncthreads();
  for (int st = 64; st > 0; st >>= 1) {
    if (t < st) red[t] += red[t + st];
    __syncthreads();
  }
  float m = red[0] / Dc;
  __syncthreads();
  float diff = val - m;
  red[t] = (t < Dc) ? diff * diff : 0.f;
  __syncthreads();
  for (int st = 64; st > 0; st >>= 1) {
    if (t < st) red[t] += red[t + st];
    __syncthreads();
  }
  float v = red[0] / Dc;
  if (t < Dc) x[row * Dc + t] = diff * rsqrtf(v + 1e-5f) * lw[t] + lb[t];
}

__global__ __launch_bounds__(64) void k_seqpool(const float* __restrict__ x,
                                                const int* __restrict__ maskb,
                                                float* __restrict__ og) {
  __shared__ float cnt_s;
  int b = blockIdx.x, d = threadIdx.x;
  if (d == 0) {
    float c = 0.f;
    for (int s = 0; s < Sc; s++) c += (float)maskb[b * Sc + s];
    cnt_s = fmaxf(c, 1.f);
  }
  __syncthreads();
  float acc = 0.f;
  for (int s = 0; s < Sc; s++)
    if (maskb[b * Sc + s]) acc += x[((size_t)b * Sc + s) * Dc + d];
  og[b * Dc + d] = acc / cnt_s;
}

__global__ __launch_bounds__(128) void k_final(const float* __restrict__ xg, const float* __restrict__ og,
                                               const float* __restrict__ fw, const float* __restrict__ fb,
                                               float* __restrict__ out) {
  __shared__ float red[128];
  int b = blockIdx.x, t = threadIdx.x;
  float acc = 0.f;
  for (int i = t; i < 2 * ODc + 2 * Dc; i += 128) {
    float cv;
    if (i < ODc)               cv = xg[b * ODc + i];
    else if (i < 2 * ODc)      cv = xg[Bc * ODc + b * ODc + (i - ODc)];
    else if (i < 2 * ODc + Dc) cv = og[b * Dc + (i - 2 * ODc)];
    else                       cv = og[Bc * Dc + b * Dc + (i - 2 * ODc - Dc)];
    acc += cv * fw[i];
  }
  red[t] = acc;
  __syncthreads();
  for (int st = 64; st > 0; st >>= 1) {
    if (t < st) red[t] += red[t + st];
    __syncthreads();
  }
  if (t == 0) out[b] = red[0] + fb[0];
}

// ======================================================================
// launch
// ======================================================================
extern "C" void kernel_launch(void* const* d_in, const int* in_sizes, int n_in,
                              void* d_out, int out_size, void* d_ws, size_t ws_size,
                              hipStream_t stream) {
  float* ws = (float*)d_ws;
  const size_t oH    = 0;
  const size_t oAGG  = oH    + (size_t)Nn * FPc;
  const size_t oDEG  = oAGG  + (size_t)Nn * FPc;
  const size_t oPS   = oDEG  + Nn;
  const size_t oPC   = oPS   + (size_t)Bc * FPc;
  const size_t oXG   = oPC   + Bc;
  const size_t oSEQ  = oXG   + (size_t)2 * Bc * ODc;
  const size_t oY    = oSEQ  + (size_t)Bc * Sc * Dc;
  const size_t oQKV  = oY    + (size_t)Bc * Sc * Dc;
  const size_t oATT  = oQKV  + (size_t)Bc * Sc * 3 * Dc;
  const size_t oOG   = oATT  + (size_t)Bc * Sc * Dc;
  const size_t oMASK = oOG   + (size_t)2 * Bc * Dc;

  float* Hbuf = ws + oH;
  float* AGG  = ws + oAGG;
  float* DEG  = ws + oDEG;
  float* PS   = ws + oPS;
  float* PC   = ws + oPC;
  float* XG   = ws + oXG;
  float* Xseq = ws + oSEQ;
  float* Yseq = ws + oY;
  float* QKV  = ws + oQKV;
  float* ATT  = ws + oATT;
  float* OG   = ws + oOG;
  int*   MSK  = (int*)(ws + oMASK);

  for (int g = 0; g < 2; g++) {
    const float* X   = (const float*)d_in[g == 0 ? 0 : 3];
    const int*   EI  = (const int*)  d_in[g == 0 ? 1 : 4];
    const int*   BAT = (const int*)  d_in[g == 0 ? 2 : 5];
    const float* GW  = (const float*)d_in[g == 0 ? 14 : 18];
    const float* GB  = (const float*)d_in[g == 0 ? 15 : 19];
    const float* FCW = (const float*)d_in[g == 0 ? 16 : 20];
    const float* FCB = (const float*)d_in[g == 0 ? 17 : 21];

    dim3 gg((Nn + 127) / 128, FPc / 128);
    k_gemm_bf16<<<gg, 256, 0, stream>>>(X, GW, Hbuf, Nn);
    k_deg_init<<<(Nn + 255) / 256, 256, 0, stream>>>(DEG, Nn);
    k_deg_edges<<<(Ec + 255) / 256, 256, 0, stream>>>(EI, DEG, Ec);
    k_dinv<<<(Nn + 255) / 256, 256, 0, stream>>>(DEG, Nn);
    k_selfbias<<<(int)(((size_t)Nn * FPc + 255) / 256), 256, 0, stream>>>(Hbuf, DEG, GB, AGG, Nn);
    k_edge_scatter<<<Ec, 256, 0, stream>>>(EI, Hbuf, DEG, AGG, Ec);
    k_pool_zero<<<(Bc * FPc + 255) / 256, 256, 0, stream>>>(PS, PC);
    k_pool<<<(Nn + 15) / 16, 256, 0, stream>>>(AGG, BAT, PS, PC, Nn);
    k_fc<<<Bc * ODc, 256, 0, stream>>>(PS, PC, FCW, FCB, XG + (size_t)g * Bc * ODc);
  }

  const float* RW  = (const float*)d_in[22];
  const float* RB  = (const float*)d_in[23];
  const float* QW  = (const float*)d_in[24];
  const float* QB  = (const float*)d_in[25];
  const float* OW  = (const float*)d_in[26];
  const float* OB  = (const float*)d_in[27];
  const float* L1W = (const float*)d_in[28];
  const float* L1B = (const float*)d_in[29];
  const float* F1W = (const float*)d_in[30];
  const float* F1B = (const float*)d_in[31];
  const float* F2W = (const float*)d_in[32];
  const float* F2B = (const float*)d_in[33];
  const float* L2W = (const float*)d_in[34];
  const float* L2B = (const float*)d_in[35];

  const int R = Bc * Sc;
  for (int s = 0; s < 2; s++) {
    const float* STR  = (const float*)d_in[6 + s * 4];
    const int*   SLEN = (const int*)  d_in[7 + s * 4];
    const float* FLP  = (const float*)d_in[8 + s * 4];
    const int*   FLEN = (const int*)  d_in[9 + s * 4];

    k_buildseq<<<R, 64, 0, stream>>>(STR, FLP, SLEN, FLEN, RW, RB, Xseq, MSK);
    for (int l = 0; l < 2; l++) {
      k_qkv<<<R, 192, 0, stream>>>(Xseq, QW + (size_t)l * 3 * Dc * Dc, QB + (size_t)l * 3 * Dc, QKV);
      k_attn<<<Bc * Hh * Sc, 256, 0, stream>>>(QKV, MSK, ATT);
      k_proj_ln<<<R, 64, 0, stream>>>(ATT, Xseq, OW + (size_t)l * Dc * Dc, OB + (size_t)l * Dc,
                                      L1W + (size_t)l * Dc, L1B + (size_t)l * Dc, Yseq);
      k_ffn_ln<<<R, 128, 0, stream>>>(Yseq, F1W + (size_t)l * FFc * Dc, F1B + (size_t)l * FFc,
                                      F2W + (size_t)l * Dc * FFc, F2B + (size_t)l * Dc,
                                      L2W + (size_t)l * Dc, L2B + (size_t)l * Dc, Xseq);
    }
    k_seqpool<<<Bc, 64, 0, stream>>>(Xseq, MSK, OG + (size_t)s * Bc * Dc);
  }

  const float* FW = (const float*)d_in[36];
  const float* FB = (const float*)d_in[37];
  k_final<<<Bc, 128, 0, stream>>>(XG, OG, FW, FB, (float*)d_out);
}
